// DecoupledMVRowColSelfAttnProcessor2_0_78194174591355
// MI455X (gfx1250) — compile-verified, matrix operand order checked by reading, not run on hardware
//
#include <hip/hip_runtime.h>

// ---------------------------------------------------------------------------
// Types for CDNA5 WMMA (wave32): 16x16x32 bf16 -> f32
// ---------------------------------------------------------------------------
typedef __bf16 bf16;
typedef __attribute__((ext_vector_type(8)))  __bf16 v8bf;
typedef __attribute__((ext_vector_type(16))) __bf16 v16bf;
typedef __attribute__((ext_vector_type(8)))  float  v8f;

#define DEVINL static __device__ __forceinline__

DEVINL v16bf make_frag(v8bf lo, v8bf hi) {
    union U { struct { v8bf lo, hi; } s; v16bf v; };
    U u; u.s.lo = lo; u.s.hi = hi; return u.v;
}

// A/B fragment loader for 16-bit 16x32 operand layout (K-major source rows).
// lane 0-15 : row = lane,    K in {0..7, 16..23}
// lane 16-31: row = lane-16, K in {8..15, 24..31}
DEVINL v16bf load_frag_bf(const bf16* base, int ld, int lane) {
    const bf16* p = base + (size_t)(lane & 15) * ld + ((lane & 16) ? 8 : 0);
    v8bf lo = *(const v8bf*)(p);
    v8bf hi = *(const v8bf*)(p + 16);
    return make_frag(lo, hi);
}

DEVINL v8f wmma_bf(v16bf a, v16bf b, v8f c) {
    // (neg_a, A, neg_b, B, c_mod, C, reuse_a, reuse_b)
    return __builtin_amdgcn_wmma_f32_16x16x32_bf16(false, a, false, b,
                                                   (short)0, c, false, false);
}

// ---------------------------------------------------------------------------
// Async Global -> LDS copy (CDNA5 ASYNCcnt path). Per-lane 16B copy:
//   GLOBAL_LOAD_ASYNC_TO_LDS_B128 vdst(lds byte addr), v[addr], off
// Low 32 bits of a generic shared pointer are the raw LDS byte offset.
// ---------------------------------------------------------------------------
DEVINL void async_copy_b128(void* lds_ptr, const void* gaddr) {
#if defined(__AMDGCN__)
    unsigned lds_off = (unsigned)(size_t)lds_ptr;
    asm volatile("global_load_async_to_lds_b128 %0, %1, off"
                 :: "v"(lds_off), "v"(gaddr)
                 : "memory");
#else
    (void)lds_ptr; (void)gaddr;
#endif
}
DEVINL void async_wait0() {
#if defined(__AMDGCN__)
    asm volatile("s_wait_asynccnt 0x0" ::: "memory");
#endif
}

// ---------------------------------------------------------------------------
// fp32 -> bf16 conversion
// ---------------------------------------------------------------------------
__global__ __launch_bounds__(256) void cvt_f32_bf16(const float* __restrict__ in,
                                                    bf16* __restrict__ out, int n) {
    int i = blockIdx.x * 256 + threadIdx.x;
    if (i < n) out[i] = (bf16)in[i];
}

// ---------------------------------------------------------------------------
// GEMM: C[M,N] = A[M,K] * W[N,K]^T  (both K-major, exactly x @ W.T)
// block = 128 thr = 4 waves; block tile = 64 rows x 256 cols.
// A K-slab (64x32) staged in LDS via async-to-LDS DMA, shared by all waves;
// each wave computes a 64x64 tile: 16 accumulators, 16 wmma per 32-K step.
// grid = (M/64, N/256)
// ---------------------------------------------------------------------------
template <bool A_FP32, bool OUT_BF16, bool ADD_BIAS, bool ACCUM_OUT, bool ADD_X>
__global__ __launch_bounds__(128) void gemm_xWT(const void* __restrict__ Ap,
                                                const bf16* __restrict__ W,
                                                const float* __restrict__ bias,
                                                const float* __restrict__ Xadd,
                                                void* __restrict__ Cp,
                                                int M, int N, int K) {
    const int lane = threadIdx.x & 31;
    const int wave = threadIdx.x >> 5;
    const int row0 = blockIdx.x * 64;
    const int col0 = blockIdx.y * 256 + wave * 64;

    __shared__ __align__(32) bf16 Alds[64][40];   // 64 rows x 32 K, padded

    v8f acc[16] = {};

    for (int kk = 0; kk < K; kk += 32) {
        __syncthreads();
        // Stage the A K-slab: 64 rows x 32 K = 256 x 16B segments.
        for (int i = threadIdx.x; i < 256; i += 128) {
            int r = i >> 2, seg = (i & 3) * 8;
            if (A_FP32) {
                const float* p = (const float*)Ap + (size_t)(row0 + r) * K + kk + seg;
                v8bf t;
#pragma unroll
                for (int e = 0; e < 8; ++e) t[e] = (bf16)p[e];
                *(v8bf*)&Alds[r][seg] = t;
            } else {
                async_copy_b128(&Alds[r][seg],
                                (const bf16*)Ap + (size_t)(row0 + r) * K + kk + seg);
            }
        }
        if (!A_FP32) async_wait0();
        __syncthreads();

        if (kk + 64 <= K)   // hint next W K-slab (emits global_prefetch_b8)
            __builtin_prefetch(W + (size_t)col0 * K + kk + 32, 0, 1);

        v16bf a[4];
#pragma unroll
        for (int i = 0; i < 4; ++i)
            a[i] = load_frag_bf(&Alds[i * 16][0], 40, lane);

#pragma unroll
        for (int j = 0; j < 4; ++j) {
            v16bf b = load_frag_bf(W + (size_t)(col0 + j * 16) * K + kk, K, lane);
#pragma unroll
            for (int i = 0; i < 4; ++i)
                acc[i * 4 + j] = wmma_bf(a[i], b, acc[i * 4 + j]);
        }
    }

    const int rhalf = (lane & 16) ? 8 : 0;
#pragma unroll
    for (int i = 0; i < 4; ++i) {
#pragma unroll
        for (int j = 0; j < 4; ++j) {
#pragma unroll
            for (int r = 0; r < 8; ++r) {
                int row = row0 + i * 16 + r + rhalf;
                int col = col0 + j * 16 + (lane & 15);
                float v = acc[i * 4 + j][r];
                if (ADD_BIAS)  v += bias[col];
                size_t idx = (size_t)row * N + col;
                if (ACCUM_OUT) v += ((const float*)Cp)[idx];
                if (ADD_X)     v += Xadd[idx];
                if (OUT_BF16) ((bf16*)Cp)[idx] = (bf16)v;
                else          ((float*)Cp)[idx] = v;
            }
        }
    }
}

// ---------------------------------------------------------------------------
// Main flash attention: per (b,h), S=1024, Dh=64.
// grid = (S/64, H, B); block = 128 (4 waves); wave -> 16 query rows.
// 32-key chunks: QK^T 4 wmma, online softmax (shfl width-16 row reductions),
// P through LDS (layout transpose), V staged transposed in LDS for PV wmma.
// ---------------------------------------------------------------------------
__global__ __launch_bounds__(128) void attn_main_kernel(const bf16* __restrict__ Q,
                                                        const bf16* __restrict__ Kt,
                                                        const bf16* __restrict__ V,
                                                        bf16* __restrict__ O) {
    const int lane = threadIdx.x & 31;
    const int wave = threadIdx.x >> 5;
    const int b = blockIdx.z, h = blockIdx.y;
    const int qbase = blockIdx.x * 64 + wave * 16;
    const int D = 1024, S = 1024;

    __shared__ __align__(32) bf16 Vt[64][40];      // [dh][key] transposed V chunk
    __shared__ __align__(32) bf16 Pl[4][16][40];   // per-wave P tile (16x32)

    const bf16* Qp = Q + ((size_t)(b * S + qbase)) * D + h * 64;
    v16bf aq0 = load_frag_bf(Qp, D, lane);
    v16bf aq1 = load_frag_bf(Qp + 32, D, lane);

    v8f o[4] = {};
    float mrow[8], lrow[8];
#pragma unroll
    for (int r = 0; r < 8; ++r) { mrow[r] = -1e30f; lrow[r] = 0.0f; }

    const float sc = 0.125f * 1.44269504088896f;   // 1/sqrt(64) * log2(e)
    const int rhalf = (lane & 16) ? 8 : 0;

    for (int c = 0; c < 32; ++c) {
        __syncthreads();
        for (int i = threadIdx.x; i < 64 * 32; i += 128) {
            int d = i >> 5, kk = i & 31;
            Vt[d][kk] = V[((size_t)(b * S + c * 32 + kk)) * D + h * 64 + d];
        }
        __syncthreads();

        const bf16* Kp = Kt + ((size_t)(b * S + c * 32)) * D + h * 64;
        v8f s0 = {}, s1 = {};
        s0 = wmma_bf(aq0, load_frag_bf(Kp, D, lane), s0);
        s0 = wmma_bf(aq1, load_frag_bf(Kp + 32, D, lane), s0);
        s1 = wmma_bf(aq0, load_frag_bf(Kp + 16 * D, D, lane), s1);
        s1 = wmma_bf(aq1, load_frag_bf(Kp + 16 * D + 32, D, lane), s1);

#pragma unroll
        for (int r = 0; r < 8; ++r) {
            float v0 = s0[r] * sc, v1 = s1[r] * sc;
            float mx = fmaxf(v0, v1);
            for (int m = 8; m; m >>= 1) mx = fmaxf(mx, __shfl_xor(mx, m, 16));
            float mn = fmaxf(mrow[r], mx);
            float alpha = exp2f(mrow[r] - mn);
            float p0 = exp2f(v0 - mn), p1 = exp2f(v1 - mn);
            float rs = p0 + p1;
            for (int m = 8; m; m >>= 1) rs += __shfl_xor(rs, m, 16);
            lrow[r] = lrow[r] * alpha + rs;
            mrow[r] = mn;
#pragma unroll
            for (int j = 0; j < 4; ++j) o[j][r] *= alpha;
            int prow = r + rhalf;
            Pl[wave][prow][lane & 15]        = (bf16)p0;
            Pl[wave][prow][16 + (lane & 15)] = (bf16)p1;
        }
        __syncthreads();

        v16bf ap = load_frag_bf(&Pl[wave][0][0], 40, lane);
#pragma unroll
        for (int j = 0; j < 4; ++j) {
            v16bf bv = load_frag_bf(&Vt[j * 16][0], 40, lane);
            o[j] = wmma_bf(ap, bv, o[j]);
        }
    }

#pragma unroll
    for (int j = 0; j < 4; ++j) {
#pragma unroll
        for (int r = 0; r < 8; ++r) {
            int qrow = qbase + r + rhalf;
            float val = o[j][r] * (1.0f / lrow[r]);
            O[((size_t)(b * S + qrow)) * D + h * 64 + j * 16 + (lane & 15)] = (bf16)val;
        }
    }
}

// ---------------------------------------------------------------------------
// Multi-view attention (row mode=0 / col mode=1), seq=128, per (vb,pos,h).
// grid = (2, H, 64); block = 128. Writes fp32 combined-view buffer:
//  row: store views 0..3 (0,2 scaled 0.5); col: add into 0,2 (0.5), store 4,5.
// ---------------------------------------------------------------------------
__global__ __launch_bounds__(128) void attn_mv_kernel(const bf16* __restrict__ Q,
                                                      const bf16* __restrict__ Kt,
                                                      const bf16* __restrict__ V,
                                                      float* __restrict__ OutF,
                                                      int mode) {
    const int lane = threadIdx.x & 31;
    const int wave = threadIdx.x >> 5;
    const int vbatch = blockIdx.z;
    const int vb = vbatch >> 5, pos = vbatch & 31;
    const int h = blockIdx.y;
    const int qbase = blockIdx.x * 64 + wave * 16;
    const int D = 1024;

    const int cview[4] = {0, 2, 4, 5};
    auto frow = [&](int j) -> int {
        if (mode == 0) return (vb * 6 + (j >> 5)) * 1024 + pos * 32 + (j & 31);
        return (vb * 6 + cview[j >> 5]) * 1024 + (j & 31) * 32 + pos;
    };
    const int ldrow = (mode == 0 ? 1 : 32) * D;   // elem stride between seq rows

    __shared__ __align__(32) bf16 Vt[64][40];
    __shared__ __align__(32) bf16 Pl[4][16][40];

    const bf16* Qp = Q + (size_t)frow(qbase) * D + h * 64;
    v16bf aq0 = load_frag_bf(Qp, ldrow, lane);
    v16bf aq1 = load_frag_bf(Qp + 32, ldrow, lane);

    v8f o[4] = {};
    float mrow[8], lrow[8];
#pragma unroll
    for (int r = 0; r < 8; ++r) { mrow[r] = -1e30f; lrow[r] = 0.0f; }

    const float sc = 0.125f * 1.44269504088896f;
    const int rhalf = (lane & 16) ? 8 : 0;

    for (int c = 0; c < 4; ++c) {
        __syncthreads();
        for (int i = threadIdx.x; i < 64 * 32; i += 128) {
            int d = i >> 5, kk = i & 31;
            Vt[d][kk] = V[(size_t)frow(c * 32 + kk) * D + h * 64 + d];
        }
        __syncthreads();

        const bf16* Kp0 = Kt + (size_t)frow(c * 32) * D + h * 64;
        const bf16* Kp1 = Kt + (size_t)frow(c * 32 + 16) * D + h * 64;
        v8f s0 = {}, s1 = {};
        s0 = wmma_bf(aq0, load_frag_bf(Kp0, ldrow, lane), s0);
        s0 = wmma_bf(aq1, load_frag_bf(Kp0 + 32, ldrow, lane), s0);
        s1 = wmma_bf(aq0, load_frag_bf(Kp1, ldrow, lane), s1);
        s1 = wmma_bf(aq1, load_frag_bf(Kp1 + 32, ldrow, lane), s1);

#pragma unroll
        for (int r = 0; r < 8; ++r) {
            float v0 = s0[r] * sc, v1 = s1[r] * sc;
            float mx = fmaxf(v0, v1);
            for (int m = 8; m; m >>= 1) mx = fmaxf(mx, __shfl_xor(mx, m, 16));
            float mn = fmaxf(mrow[r], mx);
            float alpha = exp2f(mrow[r] - mn);
            float p0 = exp2f(v0 - mn), p1 = exp2f(v1 - mn);
            float rs = p0 + p1;
            for (int m = 8; m; m >>= 1) rs += __shfl_xor(rs, m, 16);
            lrow[r] = lrow[r] * alpha + rs;
            mrow[r] = mn;
#pragma unroll
            for (int j = 0; j < 4; ++j) o[j][r] *= alpha;
            int prow = r + rhalf;
            Pl[wave][prow][lane & 15]        = (bf16)p0;
            Pl[wave][prow][16 + (lane & 15)] = (bf16)p1;
        }
        __syncthreads();

        v16bf ap = load_frag_bf(&Pl[wave][0][0], 40, lane);
#pragma unroll
        for (int j = 0; j < 4; ++j) {
            v16bf bv = load_frag_bf(&Vt[j * 16][0], 40, lane);
            o[j] = wmma_bf(ap, bv, o[j]);
        }
    }

#pragma unroll
    for (int j = 0; j < 4; ++j) {
#pragma unroll
        for (int r = 0; r < 8; ++r) {
            int qj = qbase + r + rhalf;
            int col = h * 64 + j * 16 + (lane & 15);
            float val = o[j][r] * (1.0f / lrow[r]);
            int view, s; float scale; bool addmode;
            if (mode == 0) {
                view = qj >> 5; s = pos * 32 + (qj & 31);
                scale = (view == 0 || view == 2) ? 0.5f : 1.0f;
                addmode = false;
            } else {
                int ci = qj >> 5; view = cview[ci];
                s = (qj & 31) * 32 + pos;
                scale = (ci < 2) ? 0.5f : 1.0f;
                addmode = (ci < 2);
            }
            size_t idx = ((size_t)((vb * 6 + view) * 1024 + s)) * 1024 + col;
            float wv = val * scale;
            if (addmode) wv += OutF[idx];
            OutF[idx] = wv;
        }
    }
}

// ---------------------------------------------------------------------------
// Host-side orchestration
// ---------------------------------------------------------------------------
extern "C" void kernel_launch(void* const* d_in, const int* in_sizes, int n_in,
                              void* d_out, int out_size, void* d_ws, size_t ws_size,
                              hipStream_t stream) {
    (void)in_sizes; (void)n_in; (void)out_size; (void)ws_size;
    const int D = 1024, M = 12288;                 // B*S = 12*1024
    const size_t XN = (size_t)M * D;               // 12,582,912 elems
    const size_t WN = (size_t)D * D;               // 1,048,576 elems

    const float* x    = (const float*)d_in[0];
    const float* bo   = (const float*)d_in[5];
    const float* bomv = (const float*)d_in[10];

    char* ws = (char*)d_ws;
    size_t off = 0;
    auto carve = [&](size_t bytes) { char* p = ws + off; off += (bytes + 255) & ~(size_t)255; return p; };

    bf16* xb = (bf16*)carve(XN * 2);
    bf16* wb[8];
    for (int i = 0; i < 8; ++i) wb[i] = (bf16*)carve(WN * 2);
    bf16* b1    = (bf16*)carve(XN * 2);            // q (main) then q (mv)
    bf16* b2    = (bf16*)carve(XN * 2);            // k
    bf16* b3    = (bf16*)carve(XN * 2);            // v
    bf16* attnb = (bf16*)carve(XN * 2);            // main attention output
    float* mvf  = (float*)carve(XN * 4);           // combined multi-view (fp32)

    // 1) fp32 -> bf16 conversions
    cvt_f32_bf16<<<dim3((XN + 255) / 256), dim3(256), 0, stream>>>(x, xb, (int)XN);
    const int widx[8] = {1, 2, 3, 4, 6, 7, 8, 9};  // Wq Wk Wv Wo Wq_mv Wk_mv Wv_mv Wo_mv
    for (int i = 0; i < 8; ++i)
        cvt_f32_bf16<<<dim3((WN + 255) / 256), dim3(256), 0, stream>>>(
            (const float*)d_in[widx[i]], wb[i], (int)WN);

    dim3 gemmGrid(M / 64, D / 256), gemmBlk(128);

    // 2) main QKV projections (bf16 out)
    gemm_xWT<false, true, false, false, false><<<gemmGrid, gemmBlk, 0, stream>>>(
        xb, wb[0], nullptr, nullptr, b1, M, D, D);
    gemm_xWT<false, true, false, false, false><<<gemmGrid, gemmBlk, 0, stream>>>(
        xb, wb[1], nullptr, nullptr, b2, M, D, D);
    gemm_xWT<false, true, false, false, false><<<gemmGrid, gemmBlk, 0, stream>>>(
        xb, wb[2], nullptr, nullptr, b3, M, D, D);

    // 3) main attention
    attn_main_kernel<<<dim3(16, 16, 12), dim3(128), 0, stream>>>(b1, b2, b3, attnb);

    // 4) multi-view QKV projections (reuse b1..b3)
    gemm_xWT<false, true, false, false, false><<<gemmGrid, gemmBlk, 0, stream>>>(
        xb, wb[4], nullptr, nullptr, b1, M, D, D);
    gemm_xWT<false, true, false, false, false><<<gemmGrid, gemmBlk, 0, stream>>>(
        xb, wb[5], nullptr, nullptr, b2, M, D, D);
    gemm_xWT<false, true, false, false, false><<<gemmGrid, gemmBlk, 0, stream>>>(
        xb, wb[6], nullptr, nullptr, b3, M, D, D);

    // 5) row views then column views (col read-adds into views 0,2)
    attn_mv_kernel<<<dim3(2, 16, 64), dim3(128), 0, stream>>>(b1, b2, b3, mvf, 0);
    attn_mv_kernel<<<dim3(2, 16, 64), dim3(128), 0, stream>>>(b1, b2, b3, mvf, 1);

    // 6) output projections: d_out = attn@Wo^T + bo; then += mv@Wo_mv^T + bo_mv + x
    gemm_xWT<false, false, true, false, false><<<gemmGrid, gemmBlk, 0, stream>>>(
        attnb, wb[3], bo, nullptr, d_out, M, D, D);
    gemm_xWT<true, false, true, true, true><<<gemmGrid, gemmBlk, 0, stream>>>(
        mvf, wb[7], bomv, x, d_out, M, D, D);
}
